// CustomGRU_33363305956137
// MI455X (gfx1250) — matrix operand order repaired and checked by reading, not compiled, not run on hardware
//
#include <hip/hip_runtime.h>
#include <math.h>

// ---------------------------------------------------------------------------
// CustomGRU for MI455X (gfx1250): bf16 WMMA GEMMs + persistent recurrent scan
//   L=2, DIRS=2, H=512, I=1024, B=64, T=1024
// ---------------------------------------------------------------------------

#define GB_   64
#define GT_   1024
#define GI_   1024
#define GH_   512
#define G3H_  1536
#define GM_   (GB_ * GT_)       // 65536 GEMM rows
#define OUT2H_ 1024             // 2*H

typedef __attribute__((ext_vector_type(16))) __bf16 bf16x16;
typedef __attribute__((ext_vector_type(8)))  float  f32x8;

union FragU { uint4 q[2]; bf16x16 v; };

// --- WMMA bf16 wrapper: D = A(16x32) * B(32x16) + C, f32 accumulate --------
__device__ __forceinline__ f32x8 wmma_bf16(bf16x16 a, bf16x16 b, f32x8 c) {
  return __builtin_amdgcn_wmma_f32_16x16x32_bf16(
      /*neg_a=*/false, a, /*neg_b=*/false, b,
      /*c_mod=*/(short)0, c, /*reuse_a=*/false, /*reuse_b=*/false);
}

// --- A fragment (M=16, K=32, bf16), source row-major rows=M, elem stride ld.
// ISA layout: lanes 0-15 hold K {0..7,16..23}, lanes 16-31 hold {8..15,24..31}
__device__ __forceinline__ bf16x16 load_a_frag(const __bf16* src, int ld, int lane) {
  const int row = lane & 15;
  const int ko  = (lane < 16) ? 0 : 8;
  FragU u;
  const __bf16* p = src + row * ld + ko;
  u.q[0] = *(const uint4*)(p);        // K = ko .. ko+7
  u.q[1] = *(const uint4*)(p + 16);   // K = ko+16 .. ko+23
  return u.v;
}

// --- B fragment (K=32, N=16, bf16). Memory holds W[n][k] (row n, stride ld),
// so B[k][n] = W[n][k] and each lane's 16 K-values lie in row n as two runs of 8.
// lanes 0-15: N=lane, K {0..7,16..23}; lanes 16-31: N=lane-16, K {8..15,24..31}
__device__ __forceinline__ bf16x16 load_b_frag(const __bf16* srcN, int ld, int lane) {
  const int n  = lane & 15;
  const int ko = (lane < 16) ? 0 : 8;
  FragU u;
  const __bf16* p = srcN + n * ld + ko;
  u.q[0] = *(const uint4*)(p);
  u.q[1] = *(const uint4*)(p + 16);
  return u.v;
}

// ---------------------------------------------------------------------------
// Small helper kernels
// ---------------------------------------------------------------------------
__global__ void convert_whh_kernel(const float* __restrict__ in,
                                   __bf16* __restrict__ out) {
  const size_t i = (size_t)blockIdx.x * blockDim.x + threadIdx.x; // < 2*2*1536*512
  out[i] = (__bf16)in[i];
}

__global__ void zero_bar_kernel(unsigned* __restrict__ bar) {
  const int i = blockIdx.x * blockDim.x + threadIdx.x; // < 1024
  bar[i] = 0u;
}

__global__ void init_h_kernel(const float* __restrict__ h0,
                              __bf16* __restrict__ hbuf, int layer) {
  const int i   = blockIdx.x * blockDim.x + threadIdx.x; // < 2*64*512
  const int dir = i >> 15;
  const int rem = i & 32767;
  hbuf[(size_t)dir * (GB_ * GH_) + rem] =
      (__bf16)h0[(size_t)(2 * layer + dir) * (GB_ * GH_) + rem];
}

// ---------------------------------------------------------------------------
// Input-projection GEMM:  gi[m][dir][g] = A[m][:] . W[dir][g][:] + b_ih[dir][g]
//   A: (M=65536, K=1024) fp32 (layer 0: x) or bf16 (layer 1: layer-0 output)
//   W: (DIRS, 3H, K) fp32 ; gi stored bf16 (B,T,DIRS,3H)
// Block tile 128x128, BK=64, 8 waves; wave computes 32x64 via 2x4 WMMA tiles.
// ---------------------------------------------------------------------------
__global__ __launch_bounds__(256) void gemm_gi_kernel(
    const float*  __restrict__ A32,
    const __bf16* __restrict__ A16,
    const float*  __restrict__ W,
    const float*  __restrict__ bias,
    __bf16*       __restrict__ gi) {
  const int dir = blockIdx.z;
  const int n0  = blockIdx.x * 128;
  const int m0  = blockIdx.y * 128;
  const float* Wd = W + (size_t)dir * G3H_ * GI_;

  __shared__ __bf16 As[128][72];   // 72 = 64 + 8 pad (144B rows, 16B aligned)
  __shared__ __bf16 Bs[128][72];

  const int tid  = threadIdx.x;
  const int lane = tid & 31;
  const int w    = tid >> 5;
  const int wm   = w & 3;          // row-wave (32 rows each)
  const int wn   = w >> 2;         // col-wave (64 cols each)

  f32x8 acc[2][4];
  const f32x8 zc = {0.f, 0.f, 0.f, 0.f, 0.f, 0.f, 0.f, 0.f};
#pragma unroll
  for (int i = 0; i < 2; ++i)
#pragma unroll
    for (int j = 0; j < 4; ++j) acc[i][j] = zc;

  const int ldRow = tid >> 1;          // 0..127
  const int kseg  = (tid & 1) * 32;    // 0 or 32

  for (int kc = 0; kc < GI_; kc += 64) {
    // ---- stage A tile (convert fp32 -> bf16, or straight bf16 copy) ----
    if (A32) {
      const float4* src =
          (const float4*)(A32 + (size_t)(m0 + ldRow) * GI_ + kc + kseg);
#pragma unroll
      for (int i = 0; i < 8; ++i) {
        float4 v = src[i];
        As[ldRow][kseg + 4 * i + 0] = (__bf16)v.x;
        As[ldRow][kseg + 4 * i + 1] = (__bf16)v.y;
        As[ldRow][kseg + 4 * i + 2] = (__bf16)v.z;
        As[ldRow][kseg + 4 * i + 3] = (__bf16)v.w;
      }
    } else {
      const uint4* src =
          (const uint4*)(A16 + (size_t)(m0 + ldRow) * GI_ + kc + kseg);
#pragma unroll
      for (int i = 0; i < 4; ++i)
        *(uint4*)&As[ldRow][kseg + 8 * i] = src[i];
    }
    // ---- stage B tile: rows n of W (K-contiguous), convert to bf16 ----
    {
      const float4* src =
          (const float4*)(Wd + (size_t)(n0 + ldRow) * GI_ + kc + kseg);
#pragma unroll
      for (int i = 0; i < 8; ++i) {
        float4 v = src[i];
        Bs[ldRow][kseg + 4 * i + 0] = (__bf16)v.x;
        Bs[ldRow][kseg + 4 * i + 1] = (__bf16)v.y;
        Bs[ldRow][kseg + 4 * i + 2] = (__bf16)v.z;
        Bs[ldRow][kseg + 4 * i + 3] = (__bf16)v.w;
      }
    }
    __syncthreads();

#pragma unroll
    for (int ks = 0; ks < 2; ++ks) {
      bf16x16 af0 = load_a_frag(&As[wm * 32 +  0][ks * 32], 72, lane);
      bf16x16 af1 = load_a_frag(&As[wm * 32 + 16][ks * 32], 72, lane);
#pragma unroll
      for (int nt = 0; nt < 4; ++nt) {
        bf16x16 bf_ = load_b_frag(&Bs[wn * 64 + nt * 16][ks * 32], 72, lane);
        acc[0][nt] = wmma_bf16(af0, bf_, acc[0][nt]);
        acc[1][nt] = wmma_bf16(af1, bf_, acc[1][nt]);
      }
    }
    __syncthreads();
  }

  // ---- epilogue: + b_ih, store bf16 gi at ((m*2)+dir)*1536 + n ----
#pragma unroll
  for (int nt = 0; nt < 4; ++nt) {
    const int n  = n0 + wn * 64 + nt * 16 + (lane & 15);
    const float bv = bias[dir * G3H_ + n];
#pragma unroll
    for (int mt = 0; mt < 2; ++mt) {
#pragma unroll
      for (int r = 0; r < 8; ++r) {
        const int m = m0 + wm * 32 + mt * 16 + r + ((lane < 16) ? 0 : 8);
        gi[((size_t)m * 2 + dir) * G3H_ + n] = (__bf16)(acc[mt][nt][r] + bv);
      }
    }
  }
}

// ---------------------------------------------------------------------------
// Persistent recurrent scan for one layer, both directions concurrently.
// Grid = (8 col-tiles of 64, 2 dirs) = 16 blocks, 256 threads (8 waves).
// Per step: gh = h @ w_hh^T via WMMA (fragments streamed from L2), fp32 gate
// math, bf16 h ping-pong in global, software grid barrier per timestep.
// ---------------------------------------------------------------------------
__global__ __launch_bounds__(256) void gru_scan_kernel(
    const __bf16* __restrict__ gi,    // (B,T,DIRS,3H)
    const __bf16* __restrict__ whh,   // this layer: (DIRS,3H,H) bf16
    const float*  __restrict__ bhh,   // this layer: (DIRS,3H)
    const float*  __restrict__ h0,    // (4,B,H)
    __bf16*       __restrict__ hbuf,  // (2,DIRS,B,H) ping-pong
    __bf16*       __restrict__ ybuf,  // (B,T,2H) layer-0 output (bf16)
    float*        __restrict__ fout,  // (B,T,2H) final output (fp32)
    float*        __restrict__ hn,    // (4,B,H)
    unsigned*     __restrict__ bar,   // [T] arrival counters (pre-zeroed)
    int layer) {
  const int dir  = blockIdx.y;
  const int c0   = blockIdx.x * 64;
  const int tid  = threadIdx.x;
  const int lane = tid & 31;
  const int w    = tid >> 5;
  const int mt   = w & 3;    // row-tile: b in [mt*16, mt*16+16)
  const int half = w >> 2;   // col half: cols [c0+half*32, +32)

  const int rowbase = mt * 16 + ((lane < 16) ? 0 : 8);
  const int colbase = c0 + half * 32 + (lane & 15);

  // per-lane fp32 carry of h for the (b,c) positions this lane owns
  float hprev[2][8];
  float bias[3][2];
#pragma unroll
  for (int nt = 0; nt < 2; ++nt) {
    const int c = colbase + nt * 16;
#pragma unroll
    for (int g = 0; g < 3; ++g) bias[g][nt] = bhh[dir * G3H_ + g * GH_ + c];
#pragma unroll
    for (int r = 0; r < 8; ++r)
      hprev[nt][r] =
          h0[((size_t)(2 * layer + dir) * GB_ + (rowbase + r)) * GH_ + c];
  }

  const __bf16* Wd = whh + (size_t)dir * G3H_ * GH_;
  const unsigned nb = gridDim.x * gridDim.y;
  const f32x8 zc = {0.f, 0.f, 0.f, 0.f, 0.f, 0.f, 0.f, 0.f};

  for (int s = 0; s < GT_; ++s) {
    const int t = dir ? (GT_ - 1 - s) : s;
    const __bf16* hsrc = hbuf + ((size_t)((s & 1) * 2 + dir) * GB_) * GH_;
    __bf16* hdst = hbuf + ((size_t)(((s + 1) & 1) * 2 + dir) * GB_) * GH_;

    f32x8 acc[3][2];
#pragma unroll
    for (int g = 0; g < 3; ++g)
#pragma unroll
      for (int nt = 0; nt < 2; ++nt) acc[g][nt] = zc;

    // gh tile: (16 rows b) x (2x16 cols) x 3 gates, K = H = 512
#pragma unroll 4
    for (int kc = 0; kc < GH_; kc += 32) {
      bf16x16 af = load_a_frag(hsrc + (size_t)mt * 16 * GH_ + kc, GH_, lane);
#pragma unroll
      for (int g = 0; g < 3; ++g) {
#pragma unroll
        for (int nt = 0; nt < 2; ++nt) {
          const __bf16* bp =
              Wd + (size_t)(g * GH_ + c0 + half * 32 + nt * 16) * GH_ + kc;
          acc[g][nt] = wmma_bf16(af, load_b_frag(bp, GH_, lane), acc[g][nt]);
        }
      }
    }

    // gate math in fp32
#pragma unroll
    for (int nt = 0; nt < 2; ++nt) {
      const int c = colbase + nt * 16;
#pragma unroll
      for (int r = 0; r < 8; ++r) {
        const int b = rowbase + r;
        const size_t gbase = (((size_t)b * GT_ + t) * 2 + dir) * G3H_ + c;
        const float ir  = (float)gi[gbase];
        const float iz  = (float)gi[gbase + GH_];
        const float inn = (float)gi[gbase + 2 * GH_];
        const float hr  = acc[0][nt][r] + bias[0][nt];
        const float hz  = acc[1][nt][r] + bias[1][nt];
        const float hnv = acc[2][nt][r] + bias[2][nt];
        const float rg = 1.f / (1.f + __expf(-(ir + hr)));
        const float zg = 1.f / (1.f + __expf(-(iz + hz)));
        const float ng = tanhf(inn + rg * hnv);
        const float hv = (1.f - zg) * ng + zg * hprev[nt][r];
        hprev[nt][r] = hv;

        hdst[(size_t)b * GH_ + c] = (__bf16)hv;
        const size_t oidx = ((size_t)b * GT_ + t) * OUT2H_ + dir * GH_ + c;
        if (layer == 0) ybuf[oidx] = (__bf16)hv;
        else            fout[oidx] = hv;
        if (s == GT_ - 1)
          hn[((size_t)(2 * layer + dir) * GB_ + b) * GH_ + c] = hv;
      }
    }

    // prefetch next timestep's gi row for this lane (speculative)
    if (s + 1 < GT_) {
      const int tn = dir ? (GT_ - 2 - s) : (s + 1);
      __builtin_prefetch(
          gi + (((size_t)rowbase * GT_ + tn) * 2 + dir) * G3H_ + colbase, 0, 0);
    }

    // ---- software grid barrier across the 16 blocks of this layer ----
    __threadfence();
    __syncthreads();
    if (tid == 0) {
      atomicAdd(&bar[s], 1u);
      while (__hip_atomic_load(&bar[s], __ATOMIC_RELAXED,
                               __HIP_MEMORY_SCOPE_AGENT) < nb) {
        __builtin_amdgcn_s_sleep(1);
      }
    }
    __syncthreads();
    __threadfence();
  }
}

// ---------------------------------------------------------------------------
// Host-side orchestration.
// Workspace layout (~518 MB total):
//   [0,4K)        barrier counters (T u32), re-zeroed before each layer
//   [4K, +256K)   hbuf: 2 x DIRS x B x H bf16 ping-pong
//   [+6.29M)      w_hh converted to bf16: (L,DIRS,3H,H)
//   [+402.7M)     gi: (B,T,DIRS,3H) bf16
//   [+134.2M)     ybuf: layer-0 output (B,T,2H) bf16
// ---------------------------------------------------------------------------
extern "C" void kernel_launch(void* const* d_in, const int* in_sizes, int n_in,
                              void* d_out, int out_size, void* d_ws,
                              size_t ws_size, hipStream_t stream) {
  const float* x    = (const float*)d_in[0];   // (B,T,I)
  const float* h0   = (const float*)d_in[1];   // (4,B,H)
  const float* w_ih = (const float*)d_in[2];   // (L,DIRS,3H,I)
  const float* w_hh = (const float*)d_in[3];   // (L,DIRS,3H,H)
  const float* b_ih = (const float*)d_in[4];   // (L,DIRS,3H)
  const float* b_hh = (const float*)d_in[5];   // (L,DIRS,3H)
  float* out = (float*)d_out;                  // (B,T,2H) then (4,B,H)
  float* hn  = out + (size_t)GB_ * GT_ * OUT2H_;

  char* ws = (char*)d_ws;
  unsigned* bar = (unsigned*)ws;
  __bf16* hbuf = (__bf16*)(ws + 4096);
  __bf16* whhb = (__bf16*)(ws + 4096 + 262144);
  __bf16* gi   = (__bf16*)(ws + 4096 + 262144 + 6291456);
  __bf16* ybuf = gi + (size_t)GM_ * 2 * G3H_;

  // one-time (per call) weight conversion: 2*2*1536*512 = 3,145,728 elems
  convert_whh_kernel<<<12288, 256, 0, stream>>>(w_hh, whhb);

  for (int l = 0; l < 2; ++l) {
    zero_bar_kernel<<<4, 256, 0, stream>>>(bar);
    init_h_kernel<<<256, 256, 0, stream>>>(h0, hbuf, l);

    if (l == 0) {
      gemm_gi_kernel<<<dim3(12, 512, 2), 256, 0, stream>>>(
          x, (const __bf16*)nullptr, w_ih, b_ih, gi);
    } else {
      gemm_gi_kernel<<<dim3(12, 512, 2), 256, 0, stream>>>(
          (const float*)nullptr, ybuf, w_ih + (size_t)2 * G3H_ * GI_,
          b_ih + 2 * G3H_, gi);
    }

    gru_scan_kernel<<<dim3(8, 2), 256, 0, stream>>>(
        gi, whhb + (size_t)l * 2 * G3H_ * GH_, b_hh + l * 2 * G3H_, h0, hbuf,
        ybuf, out, hn, bar, l);
  }
}
